// RBF_40492951667428
// MI455X (gfx1250) — matrix operand order repaired and checked by reading, not compiled, not run on hardware
//
#include <hip/hip_runtime.h>

typedef __attribute__((ext_vector_type(16))) __bf16 v16bf;
typedef __attribute__((ext_vector_type(8)))  float  v8f;

#define NROWS 8192
#define KDIM  512
#define KC    64
#define LDSTR 72          // 64 + 8 bf16 pad: 144 B/row, 16B-aligned, conflict-skewed
#define TSTR  132         // transpose tile row stride (floats): 528 B, 16B-aligned

union FragAB {
  v16bf v;
  uint4 q[2];
};

// ---- CDNA5 async global->LDS (ASYNCcnt) helpers -----------------------------
// GVS mode: uniform 64-bit base in an SGPR pair, 32-bit per-lane byte offset.
__device__ __forceinline__ void async_load_b128(unsigned lds_addr, unsigned voff,
                                                unsigned long long sbase) {
  asm volatile("global_load_async_to_lds_b128 %0, %1, %2"
               :: "v"(lds_addr), "v"(voff), "s"(sbase)
               : "memory");
}
__device__ __forceinline__ void wait_async0() {
  asm volatile("s_wait_asynccnt 0x0" ::: "memory");
}

// ---------------------------------------------------------------------------
// Kernel 1 (fast path): Xb = bf16(X); sq[i] = sum_k bf16(x[i,k])^2 in fp32.
// Using bf16-rounded values keeps diag(d2) consistent with the bf16 gram.
// ---------------------------------------------------------------------------
__global__ __launch_bounds__(256) void rbf_cvt_sq_kernel(const float* __restrict__ X,
                                                         __bf16* __restrict__ Xb,
                                                         float* __restrict__ sq) {
  const int row  = blockIdx.x * 8 + (threadIdx.x >> 5);  // one wave32 per row
  const int lane = threadIdx.x & 31;
  const float4* src = (const float4*)(X + (size_t)row * KDIM);
  uint2* dst = (uint2*)(Xb + (size_t)row * KDIM);
  float s = 0.0f;
  #pragma unroll
  for (int it = 0; it < 4; ++it) {
    const int idx = lane + 32 * it;              // 128 float4 per row
    const float4 f = src[idx];
    union { __bf16 b[4]; uint2 u; } p;
    p.b[0] = (__bf16)f.x; p.b[1] = (__bf16)f.y;
    p.b[2] = (__bf16)f.z; p.b[3] = (__bf16)f.w;
    dst[idx] = p.u;
    const float x0 = (float)p.b[0], x1 = (float)p.b[1];
    const float x2 = (float)p.b[2], x3 = (float)p.b[3];
    s += x0 * x0 + x1 * x1 + x2 * x2 + x3 * x3;
  }
  #pragma unroll
  for (int off = 16; off > 0; off >>= 1)
    s += __shfl_xor(s, off, 32);
  if (lane == 0) sq[row] = s;
}

// ---------------------------------------------------------------------------
// Kernel 2 (fast path): symmetric RBF GEMM. 2080 lower-triangle 128x128 blocks.
// bf16 panels staged with global_load_async_to_lds_b128 (GVS), double-buffered.
// Mirror tile written through an LDS transpose buffer (skipped on diagonal).
// ---------------------------------------------------------------------------
__global__ __launch_bounds__(256) void rbf_gemm_sym(const __bf16* __restrict__ Xb,
                                                    const float* __restrict__ sq,
                                                    float* __restrict__ out) {
  // 4 panels (A0,B0,A1,B1) of 128*LDSTR bf16 = 73728 B; reused as 128*TSTR f32
  // transpose tile (67584 B) after the K loop.
  __shared__ __align__(16) unsigned char smem[4 * 128 * LDSTR * 2];
  __bf16* pan = (__bf16*)smem;
  float*  T   = (float*)smem;

  const int tid   = threadIdx.x;
  const int lane  = tid & 31;
  const int wave  = tid >> 5;
  const int waveM = wave >> 2;   // 0..1 -> 64-row slab
  const int waveN = wave & 3;    // 0..3 -> 32-col slab
  const int half  = lane >> 4;
  const int l16   = lane & 15;

  // Decode lower-triangle block (bi >= bj) from linear blockIdx.x.
  const int t = blockIdx.x;
  int bi = (int)((sqrtf(8.0f * (float)t + 1.0f) - 1.0f) * 0.5f);
  while ((bi + 1) * (bi + 2) / 2 <= t) ++bi;
  while (bi * (bi + 1) / 2 > t) --bi;
  const int bj = t - bi * (bi + 1) / 2;
  const int rowBase = bi * 128;
  const int colBase = bj * 128;

  const unsigned long long xbase = (unsigned long long)(uintptr_t)Xb;

  auto stage = [&](int buf, int kc0) {
    __bf16* lA = pan + buf * (2 * 128 * LDSTR);
    __bf16* lB = lA + 128 * LDSTR;
    #pragma unroll
    for (int it = 0; it < 4; ++it) {
      const int j  = tid + 256 * it;      // 0..1023: 128 rows x 8 x b128
      const int r  = j >> 3;
      const int c8 = (j & 7) * 8;         // bf16 column offset, 16B granules
      const unsigned offA =
          ((unsigned)(rowBase + r) * KDIM + (unsigned)(kc0 + c8)) * 2u;
      const unsigned offB =
          ((unsigned)(colBase + r) * KDIM + (unsigned)(kc0 + c8)) * 2u;
      async_load_b128((unsigned)(uintptr_t)&lA[r * LDSTR + c8], offA, xbase);
      async_load_b128((unsigned)(uintptr_t)&lB[r * LDSTR + c8], offB, xbase);
    }
  };

  v8f acc[4][2] = {};

  stage(0, 0);
  wait_async0();
  __syncthreads();

  for (int kc = 0; kc < KDIM / KC; ++kc) {
    const int cur = kc & 1;
    if (kc + 1 < KDIM / KC) stage(cur ^ 1, (kc + 1) * KC);

    const __bf16* lA = pan + cur * (2 * 128 * LDSTR);
    const __bf16* lB = lA + 128 * LDSTR;

    #pragma unroll
    for (int ks = 0; ks < KC; ks += 32) {
      FragAB a[4], b[2];
      // A (16x32 bf16): lane l16 = row M; elems 0-7: K = half*8 + e,
      // elems 8-15: K = 16 + half*8 + (e-8).
      #pragma unroll
      for (int mt = 0; mt < 4; ++mt) {
        const __bf16* base = &lA[(waveM * 64 + mt * 16 + l16) * LDSTR + ks];
        a[mt].q[0] = *(const uint4*)(base + half * 8);
        a[mt].q[1] = *(const uint4*)(base + 16 + half * 8);
      }
      // B (32x16 bf16): lane l16 = column N; elems 0-7: K = half*8 + e,
      // elems 8-15: K = 16 + half*8 + (e-8).
      #pragma unroll
      for (int nt = 0; nt < 2; ++nt) {
        const __bf16* base = &lB[(waveN * 32 + nt * 16 + l16) * LDSTR + ks];
        b[nt].q[0] = *(const uint4*)(base + half * 8);
        b[nt].q[1] = *(const uint4*)(base + 16 + half * 8);
      }
      #pragma unroll
      for (int mt = 0; mt < 4; ++mt)
        #pragma unroll
        for (int nt = 0; nt < 2; ++nt)
          acc[mt][nt] = __builtin_amdgcn_wmma_f32_16x16x32_bf16(
              false, a[mt].v, false, b[nt].v, (short)0, acc[mt][nt], false, false);
    }

    if (kc + 1 < KDIM / KC) {
      wait_async0();
      __syncthreads();
    }
  }

  __syncthreads();  // panels dead; smem becomes the transpose tile

  // Epilogue: direct store of the lower tile + stash transposed copy in LDS.
  #pragma unroll
  for (int nt = 0; nt < 2; ++nt) {
    const int cl  = waveN * 32 + nt * 16 + l16;
    const int col = colBase + cl;
    const float sqc = sq[col];
    #pragma unroll
    for (int mt = 0; mt < 4; ++mt) {
      #pragma unroll
      for (int r = 0; r < 8; ++r) {
        const int rl  = waveM * 64 + mt * 16 + half * 8 + r;
        const int row = rowBase + rl;
        float d2 = fmaxf(sq[row] + sqc - 2.0f * acc[mt][nt][r], 0.0f);
        const float v = __expf(-0.5f * d2);   // SIGMA=1 -> 1/(2s^2)=0.5
        out[(size_t)row * NROWS + col] = v;
        T[cl * TSTR + rl] = v;
      }
    }
  }

  if (bi != bj) {   // mirror the tile to the upper triangle, coalesced
    __syncthreads();
    #pragma unroll
    for (int i = tid * 4; i < 128 * 128; i += 256 * 4) {
      const int r = i >> 7;
      const int c = i & 127;
      const float4 v4 = *(const float4*)&T[r * TSTR + c];
      *(float4*)&out[(size_t)(colBase + r) * NROWS + rowBase + c] = v4;
    }
  }
}

// ---------------------------------------------------------------------------
// Fallback path (small workspace): fp32-reading kernels, full 64x64 grid.
// ---------------------------------------------------------------------------
__global__ __launch_bounds__(256) void rbf_sq_kernel(const float* __restrict__ X,
                                                     float* __restrict__ sq) {
  const int row  = blockIdx.x * 8 + (threadIdx.x >> 5);
  const int lane = threadIdx.x & 31;
  const float* xr = X + (size_t)row * KDIM;
  float s = 0.0f;
  for (int k = lane; k < KDIM; k += 32) {
    float v = (float)(__bf16)xr[k];
    s += v * v;
  }
  #pragma unroll
  for (int off = 16; off > 0; off >>= 1)
    s += __shfl_xor(s, off, 32);
  if (lane == 0) sq[row] = s;
}

__global__ __launch_bounds__(256) void rbf_gemm_kernel(const float* __restrict__ X,
                                                       const float* __restrict__ sq,
                                                       float* __restrict__ out) {
  __shared__ __align__(16) __bf16 lA[128 * LDSTR];
  __shared__ __align__(16) __bf16 lB[128 * LDSTR];

  const int tid   = threadIdx.x;
  const int lane  = tid & 31;
  const int wave  = tid >> 5;
  const int waveM = wave >> 2;
  const int waveN = wave & 3;
  const int half  = lane >> 4;
  const int l16   = lane & 15;

  const int rowBase = blockIdx.y * 128;
  const int colBase = blockIdx.x * 128;

  v8f acc[4][2] = {};

  for (int kc0 = 0; kc0 < KDIM; kc0 += KC) {
    __syncthreads();
    #pragma unroll
    for (int i = tid * 4; i < 128 * KC; i += 256 * 4) {
      const int r = i / KC;
      const int c = i % KC;
      const float4 a4 = *(const float4*)(X + (size_t)(rowBase + r) * KDIM + kc0 + c);
      const float4 b4 = *(const float4*)(X + (size_t)(colBase + r) * KDIM + kc0 + c);
      __bf16* pa = &lA[r * LDSTR + c];
      __bf16* pb = &lB[r * LDSTR + c];
      pa[0] = (__bf16)a4.x; pa[1] = (__bf16)a4.y; pa[2] = (__bf16)a4.z; pa[3] = (__bf16)a4.w;
      pb[0] = (__bf16)b4.x; pb[1] = (__bf16)b4.y; pb[2] = (__bf16)b4.z; pb[3] = (__bf16)b4.w;
    }
    __syncthreads();

    #pragma unroll
    for (int ks = 0; ks < KC; ks += 32) {
      FragAB a[4], b[2];
      #pragma unroll
      for (int mt = 0; mt < 4; ++mt) {
        const __bf16* base = &lA[(waveM * 64 + mt * 16 + l16) * LDSTR + ks];
        a[mt].q[0] = *(const uint4*)(base + half * 8);
        a[mt].q[1] = *(const uint4*)(base + 16 + half * 8);
      }
      #pragma unroll
      for (int nt = 0; nt < 2; ++nt) {
        const __bf16* base = &lB[(waveN * 32 + nt * 16 + l16) * LDSTR + ks];
        b[nt].q[0] = *(const uint4*)(base + half * 8);
        b[nt].q[1] = *(const uint4*)(base + 16 + half * 8);
      }
      #pragma unroll
      for (int mt = 0; mt < 4; ++mt)
        #pragma unroll
        for (int nt = 0; nt < 2; ++nt)
          acc[mt][nt] = __builtin_amdgcn_wmma_f32_16x16x32_bf16(
              false, a[mt].v, false, b[nt].v, (short)0, acc[mt][nt], false, false);
    }
  }

  #pragma unroll
  for (int nt = 0; nt < 2; ++nt) {
    const int col = colBase + waveN * 32 + nt * 16 + l16;
    const float sqc = sq[col];
    #pragma unroll
    for (int mt = 0; mt < 4; ++mt) {
      #pragma unroll
      for (int r = 0; r < 8; ++r) {
        const int row = rowBase + waveM * 64 + mt * 16 + half * 8 + r;
        float d2 = fmaxf(sq[row] + sqc - 2.0f * acc[mt][nt][r], 0.0f);
        out[(size_t)row * NROWS + col] = __expf(-0.5f * d2);
      }
    }
  }
}

extern "C" void kernel_launch(void* const* d_in, const int* in_sizes, int n_in,
                              void* d_out, int out_size, void* d_ws, size_t ws_size,
                              hipStream_t stream) {
  const float* X = (const float*)d_in[0];
  float* out     = (float*)d_out;

  const size_t XB_BYTES = (size_t)NROWS * KDIM * sizeof(__bf16);   // 8 MB
  const size_t NEED     = XB_BYTES + (size_t)NROWS * sizeof(float);

  if (ws_size >= NEED) {
    __bf16* Xb = (__bf16*)d_ws;
    float*  sq = (float*)((char*)d_ws + XB_BYTES);
    rbf_cvt_sq_kernel<<<NROWS / 8, 256, 0, stream>>>(X, Xb, sq);
    const int nblk = (NROWS / 128) * (NROWS / 128 + 1) / 2;        // 2080
    rbf_gemm_sym<<<nblk, 256, 0, stream>>>(Xb, sq, out);
  } else {
    float* sq = (float*)d_ws;                                      // 32 KB
    rbf_sq_kernel<<<NROWS / 8, 256, 0, stream>>>(X, sq);
    rbf_gemm_kernel<<<dim3(NROWS / 128, NROWS / 128), 256, 0, stream>>>(X, sq, out);
  }
}
